// MSDeformAttn_6296422056348
// MI455X (gfx1250) — matrix operand order repaired and checked by reading, not compiled, not run on hardware
//
#include <hip/hip_runtime.h>

typedef __attribute__((ext_vector_type(16))) _Float16 v16h;
typedef __attribute__((ext_vector_type(8)))  _Float16 v8h;
typedef __attribute__((ext_vector_type(8)))  float    v8f;

#define D_MODEL  96
#define N_HEADS  6
#define N_LEVELS 4
#define N_POINTS 4
#define D_HEAD   16
#define LEN_IN   21760
#define LEN_Q    21760
#define N_BF     8          // N_BATCH * N_FRAMES
#define N_OFF    288        // 6*4*4*3
#define N_ATT    96         // 6*4*4
#define MTILES   ((N_BF * LEN_Q) / 16)   // 10880

// Packed-weight fragment table (f16, WMMA B layout), fragment = 32 lanes x 16 halfs = 1KB
#define FRAG_WV   0
#define FRAG_WOFF 18
#define FRAG_WATT 72
#define FRAG_WO   90
#define N_FRAGS   108
#define VALUE_BYTES ((size_t)N_BF * LEN_IN * D_MODEL * 2)   // f16 value: 33,423,360 (32B-aligned)

// ---- WMMA fragment helpers -------------------------------------------------
// A from fp32 memory: 16x32 (MxK) f16 layout:
//   lane m=lane&15; K = kbase + (lane<16?0:8) + (e<8? e : 8+e)
__device__ __forceinline__ v16h load_a_frag(const float* row_base, int ldm, int kbase, int lane) {
    int m    = lane & 15;
    int koff = (lane < 16) ? 0 : 8;
    const float* p = row_base + (size_t)m * ldm + kbase + koff;
    v16h a;
#pragma unroll
    for (int e = 0; e < 8; ++e) a[e] = (_Float16)p[e];
#pragma unroll
    for (int e = 0; e < 8; ++e) a[8 + e] = (_Float16)p[16 + e];
    return a;
}

// A from f16 LDS (16B-aligned rows): two 16-byte vector loads per lane
__device__ __forceinline__ v16h load_a_frag_h(const _Float16* row_base, int ldm, int kbase, int lane) {
    int m    = lane & 15;
    int koff = (lane < 16) ? 0 : 8;
    const _Float16* p = row_base + (size_t)m * ldm + kbase + koff;
    v8h lo = *(const v8h*)(p);
    v8h hi = *(const v8h*)(p + 16);
    v16h a;
#pragma unroll
    for (int e = 0; e < 8; ++e) { a[e] = lo[e]; a[8 + e] = hi[e]; }
    return a;
}

// B from fp32 weights (pack kernel only): N = ntile*16 + (lane&15); K = kbase + (lane<16?0:8) + (e<8? e : 8+e)
__device__ __forceinline__ v16h load_b_frag(const float* W, int N, int ntile, int kbase, int lane) {
    int n  = ntile * 16 + (lane & 15);
    int kb = kbase + ((lane < 16) ? 0 : 8);
    const float* p = W + (size_t)kb * N + n;
    v16h b;
#pragma unroll
    for (int e = 0; e < 16; ++e) b[e] = (_Float16)p[(size_t)(e < 8 ? e : e + 8) * N];
    return b;
}

// B fragment from the packed table: one contiguous 32B load per lane
__device__ __forceinline__ v16h load_pk(const _Float16* pk, int frag, int lane) {
    return *(const v16h*)(pk + (((size_t)frag * 32 + lane) << 4));
}

// ---- Kernel 0: pack all weights into f16 B-fragment layout (runs once) -----
__global__ void pack_weights_kernel(const float* __restrict__ Wv,
                                    const float* __restrict__ Woff,
                                    const float* __restrict__ Watt,
                                    const float* __restrict__ Wo,
                                    _Float16* __restrict__ pk) {
    int frag = blockIdx.x;          // 0..107, one wave per fragment
    int lane = threadIdx.x;         // 32 threads
    const float* W;
    int N, local;
    if (frag < FRAG_WOFF)      { W = Wv;   N = D_MODEL; local = frag; }
    else if (frag < FRAG_WATT) { W = Woff; N = N_OFF;   local = frag - FRAG_WOFF; }
    else if (frag < FRAG_WO)   { W = Watt; N = N_ATT;   local = frag - FRAG_WATT; }
    else                       { W = Wo;   N = D_MODEL; local = frag - FRAG_WO; }
    int nt = local / 3, kc = local % 3;
    v16h b = load_b_frag(W, N, nt, kc * 32, lane);
    *(v16h*)(pk + (((size_t)frag * 32 + lane) << 4)) = b;
}

// ---- Kernel 1: value = input_flatten @ W_v + b_v (WMMA) -> f16 value -------
__global__ void __launch_bounds__(64)
value_proj_kernel(const float* __restrict__ X, const _Float16* __restrict__ pk,
                  const float* __restrict__ bias, _Float16* __restrict__ Y) {
    int mtile = (blockIdx.x * blockDim.x + threadIdx.x) >> 5;
    int lane  = threadIdx.x & 31;
    const float* arow = X + (size_t)mtile * 16 * D_MODEL;

    v16h a0 = load_a_frag(arow, D_MODEL, 0,  lane);
    v16h a1 = load_a_frag(arow, D_MODEL, 32, lane);
    v16h a2 = load_a_frag(arow, D_MODEL, 64, lane);

    int col16 = lane & 15;
    int rbase = mtile * 16 + ((lane < 16) ? 0 : 8);
#pragma unroll
    for (int nt = 0; nt < D_MODEL / 16; ++nt) {
        v8f c = {};
        c = __builtin_amdgcn_wmma_f32_16x16x32_f16(false, a0, false, load_pk(pk, FRAG_WV + nt * 3 + 0, lane), (short)0, c, false, false);
        c = __builtin_amdgcn_wmma_f32_16x16x32_f16(false, a1, false, load_pk(pk, FRAG_WV + nt * 3 + 1, lane), (short)0, c, false, false);
        c = __builtin_amdgcn_wmma_f32_16x16x32_f16(false, a2, false, load_pk(pk, FRAG_WV + nt * 3 + 2, lane), (short)0, c, false, false);
        int col  = nt * 16 + col16;
        float bv = bias[col];
#pragma unroll
        for (int r = 0; r < 8; ++r)
            Y[(size_t)(rbase + r) * D_MODEL + col] = (_Float16)(c[r] + bv);
    }
}

// ---- Kernel 2: fused soff/att WMMA + softmax + trilinear sample + W_o WMMA -
struct WaveLDS {
    _Float16 soff[16][N_OFF];    // 9.0 KB
    _Float16 att[16][N_ATT];     // 3.0 KB
    _Float16 mid[16][D_MODEL];   // 3.0 KB (16B-aligned base: 12288 into struct)
    float    rpq[16][12];        // 0.75 KB reference points cache
};

__global__ void __launch_bounds__(64)
msda_fused_kernel(const float* __restrict__ query,
                  const float* __restrict__ refp,
                  const _Float16* __restrict__ value,
                  const _Float16* __restrict__ pk,
                  const float* __restrict__ b_off,
                  const float* __restrict__ b_att,
                  const float* __restrict__ b_o,
                  float* __restrict__ out) {
    __shared__ WaveLDS lds[2];
    int wid  = threadIdx.x >> 5;
    int lane = threadIdx.x & 31;
    WaveLDS& L = lds[wid];

    int tile  = blockIdx.x * 2 + wid;     // 0..MTILES-1
    int bf    = tile / (LEN_Q / 16);
    int qt    = tile % (LEN_Q / 16);
    int q0    = qt * 16;
    int batch = bf / 4;                   // bf = batch*nf + frame

    // cache reference points for the 16 queries (12 floats each)
#pragma unroll
    for (int i = 0; i < 6; ++i) {
        int idx = lane + 32 * i;          // 0..191
        L.rpq[idx / 12][idx % 12] =
            refp[((size_t)batch * LEN_Q + q0 + idx / 12) * 12 + (idx % 12)];
    }

    const float* qrow = query + ((size_t)bf * LEN_Q + q0) * D_MODEL;
    v16h a0 = load_a_frag(qrow, D_MODEL, 0,  lane);
    v16h a1 = load_a_frag(qrow, D_MODEL, 32, lane);
    v16h a2 = load_a_frag(qrow, D_MODEL, 64, lane);

    int col16 = lane & 15;
    int rbase = (lane < 16) ? 0 : 8;

    // --- sampling offsets: soff = q @ W_off + b_off  -> LDS (f16) ---
    for (int nt = 0; nt < N_OFF / 16; ++nt) {
        v8f c = {};
        c = __builtin_amdgcn_wmma_f32_16x16x32_f16(false, a0, false, load_pk(pk, FRAG_WOFF + nt * 3 + 0, lane), (short)0, c, false, false);
        c = __builtin_amdgcn_wmma_f32_16x16x32_f16(false, a1, false, load_pk(pk, FRAG_WOFF + nt * 3 + 1, lane), (short)0, c, false, false);
        c = __builtin_amdgcn_wmma_f32_16x16x32_f16(false, a2, false, load_pk(pk, FRAG_WOFF + nt * 3 + 2, lane), (short)0, c, false, false);
        float bv = b_off[nt * 16 + col16];
#pragma unroll
        for (int r = 0; r < 8; ++r) L.soff[rbase + r][nt * 16 + col16] = (_Float16)(c[r] + bv);
    }
    // --- attention logits: q @ W_att + b_att -> LDS (f16) ---
#pragma unroll
    for (int nt = 0; nt < N_ATT / 16; ++nt) {
        v8f c = {};
        c = __builtin_amdgcn_wmma_f32_16x16x32_f16(false, a0, false, load_pk(pk, FRAG_WATT + nt * 3 + 0, lane), (short)0, c, false, false);
        c = __builtin_amdgcn_wmma_f32_16x16x32_f16(false, a1, false, load_pk(pk, FRAG_WATT + nt * 3 + 1, lane), (short)0, c, false, false);
        c = __builtin_amdgcn_wmma_f32_16x16x32_f16(false, a2, false, load_pk(pk, FRAG_WATT + nt * 3 + 2, lane), (short)0, c, false, false);
        float bv = b_att[nt * 16 + col16];
#pragma unroll
        for (int r = 0; r < 8; ++r) L.att[rbase + r][nt * 16 + col16] = (_Float16)(c[r] + bv);
    }
    __syncthreads();

    // --- softmax over 16 (level,point) logits per (q,h): 96 pairs, 3 per lane ---
#pragma unroll
    for (int i = 0; i < 3; ++i) {
        int pair = lane + 32 * i;
        int q = pair / N_HEADS, h = pair % N_HEADS;
        _Float16* a = &L.att[q][h * 16];
        float v[16];
#pragma unroll
        for (int j = 0; j < 16; ++j) v[j] = (float)a[j];
        float mx = v[0];
#pragma unroll
        for (int j = 1; j < 16; ++j) mx = fmaxf(mx, v[j]);
        float s = 0.f;
#pragma unroll
        for (int j = 0; j < 16; ++j) { v[j] = __expf(v[j] - mx); s += v[j]; }
        float inv = 1.f / s;
#pragma unroll
        for (int j = 0; j < 16; ++j) a[j] = (_Float16)(v[j] * inv);
    }
    __syncthreads();

    // --- trilinear deformable sampling: half-wave (16 lanes = 16 channels) per (q,h) ---
    const int LS_[4] = {0, 16384, 20480, 21504};
    const int LW_[4] = {64, 32, 16, 8};
    const int LH_[4] = {64, 32, 16, 8};
    const int LD_[4] = {4, 4, 4, 4};

    int d    = lane & 15;
    int half = lane >> 4;
    for (int u = 0; u < 48; ++u) {
        int unit = u * 2 + half;          // 96 (q,h) units, 2 per iteration
        int q = unit / N_HEADS;
        int h = unit % N_HEADS;
        float acc = 0.f;
#pragma unroll
        for (int l = 0; l < N_LEVELS; ++l) {
            const int W = LW_[l], H = LH_[l], Dd = LD_[l];
            const float invW = 1.f / (float)W, invH = 1.f / (float)H, invD = 1.f / (float)Dd;
            const _Float16* vbase = value + ((size_t)bf * LEN_IN + LS_[l]) * D_MODEL + h * D_HEAD + d;
            float rx = L.rpq[q][l * 3 + 0], ry = L.rpq[q][l * 3 + 1], rz = L.rpq[q][l * 3 + 2];
#pragma unroll
            for (int p = 0; p < N_POINTS; ++p) {
                int sb = ((h * N_LEVELS + l) * N_POINTS + p) * 3;
                float lx = rx + (float)L.soff[q][sb + 0] * invW;
                float ly = ry + (float)L.soff[q][sb + 1] * invH;
                float lz = rz + (float)L.soff[q][sb + 2] * invD;
                float aw = (float)L.att[q][h * 16 + l * 4 + p];
                float x = lx * (float)W - 0.5f, y = ly * (float)H - 0.5f, z = lz * (float)Dd - 0.5f;
                float xf = floorf(x), yf = floorf(y), zf = floorf(z);
                int   x0 = (int)xf,  y0 = (int)yf,  z0 = (int)zf;
                float dx = x - xf,   dy = y - yf,   dz = z - zf;

                // per-axis clamp + masked weights -> all 8 corner loads unconditional
                int cx[2], cy[2], cz[2];
                float wx[2], wy[2], wz[2];
                int x1 = x0 + 1, y1 = y0 + 1, z1 = z0 + 1;
                cx[0] = x0 < 0 ? 0 : (x0 >= W  ? W  - 1 : x0);
                cx[1] = x1 < 0 ? 0 : (x1 >= W  ? W  - 1 : x1);
                cy[0] = y0 < 0 ? 0 : (y0 >= H  ? H  - 1 : y0);
                cy[1] = y1 < 0 ? 0 : (y1 >= H  ? H  - 1 : y1);
                cz[0] = z0 < 0 ? 0 : (z0 >= Dd ? Dd - 1 : z0);
                cz[1] = z1 < 0 ? 0 : (z1 >= Dd ? Dd - 1 : z1);
                wx[0] = (x0 >= 0 && x0 < W)  ? (1.f - dx) : 0.f;
                wx[1] = (x1 >= 0 && x1 < W)  ? dx         : 0.f;
                wy[0] = (y0 >= 0 && y0 < H)  ? (1.f - dy) : 0.f;
                wy[1] = (y1 >= 0 && y1 < H)  ? dy         : 0.f;
                wz[0] = (z0 >= 0 && z0 < Dd) ? (1.f - dz) : 0.f;
                wz[1] = (z1 >= 0 && z1 < Dd) ? dz         : 0.f;

                float s = 0.f;
#pragma unroll
                for (int k = 0; k < 2; ++k)
#pragma unroll
                    for (int j = 0; j < 2; ++j) {
                        int rowb = (cz[k] * H + cy[j]) * W;
                        float wyz = wz[k] * wy[j];
#pragma unroll
                        for (int i = 0; i < 2; ++i) {
                            float g = (float)vbase[(size_t)(rowb + cx[i]) * D_MODEL];
                            s += (wyz * wx[i]) * g;
                        }
                    }
                acc += aw * s;
            }
        }
        L.mid[q][h * D_HEAD + d] = (_Float16)acc;
    }
    __syncthreads();

    // --- final projection: out = mid @ W_o + b_o (WMMA, A from f16 LDS) ---
    const _Float16* mrow = &L.mid[0][0];
    v16h m0 = load_a_frag_h(mrow, D_MODEL, 0,  lane);
    v16h m1 = load_a_frag_h(mrow, D_MODEL, 32, lane);
    v16h m2 = load_a_frag_h(mrow, D_MODEL, 64, lane);
#pragma unroll
    for (int nt = 0; nt < D_MODEL / 16; ++nt) {
        v8f c = {};
        c = __builtin_amdgcn_wmma_f32_16x16x32_f16(false, m0, false, load_pk(pk, FRAG_WO + nt * 3 + 0, lane), (short)0, c, false, false);
        c = __builtin_amdgcn_wmma_f32_16x16x32_f16(false, m1, false, load_pk(pk, FRAG_WO + nt * 3 + 1, lane), (short)0, c, false, false);
        c = __builtin_amdgcn_wmma_f32_16x16x32_f16(false, m2, false, load_pk(pk, FRAG_WO + nt * 3 + 2, lane), (short)0, c, false, false);
        float bv = b_o[nt * 16 + col16];
#pragma unroll
        for (int r = 0; r < 8; ++r)
            out[((size_t)bf * LEN_Q + q0 + rbase + r) * D_MODEL + nt * 16 + col16] = c[r] + bv;
    }
}

extern "C" void kernel_launch(void* const* d_in, const int* in_sizes, int n_in,
                              void* d_out, int out_size, void* d_ws, size_t ws_size,
                              hipStream_t stream) {
    const float* query = (const float*)d_in[0];
    const float* refp  = (const float*)d_in[1];
    const float* inpf  = (const float*)d_in[2];
    // d_in[3], d_in[4]: level shapes/starts -- compile-time constants here
    const float* W_off = (const float*)d_in[5];
    const float* b_off = (const float*)d_in[6];
    const float* W_att = (const float*)d_in[7];
    const float* b_att = (const float*)d_in[8];
    const float* W_v   = (const float*)d_in[9];
    const float* b_v   = (const float*)d_in[10];
    const float* W_o   = (const float*)d_in[11];
    const float* b_o   = (const float*)d_in[12];
    float* out = (float*)d_out;

    _Float16* value = (_Float16*)d_ws;                           // 33.4 MB f16
    _Float16* pk    = (_Float16*)((char*)d_ws + VALUE_BYTES);    // 108 KB packed fragments

    // Kernel 0: pack weights into f16 WMMA-B fragment layout (one wave per fragment).
    pack_weights_kernel<<<N_FRAGS, 32, 0, stream>>>(W_v, W_off, W_att, W_o, pk);

    // Kernel 1: value projection. One wave per M-tile (A loaded once for 6 N-tiles).
    value_proj_kernel<<<MTILES / 2, 64, 0, stream>>>(inpf, pk, b_v, value);

    // Kernel 2: fused projections + sampling + output projection. One wave per 16-query tile.
    msda_fused_kernel<<<MTILES / 2, 64, 0, stream>>>(query, refp, value, pk,
                                                     b_off, b_att, b_o, out);
}
